// SAttention_34935263985838
// MI455X (gfx1250) — matrix operand order repaired and checked by reading, not compiled, not run on hardware
//
#include <hip/hip_runtime.h>

// Problem constants (from reference)
constexpr int S  = 2048;
constexpr int NB = 4;
constexpr int D  = 1024;
constexpr int H  = 16;
constexpr int DH = 64;          // head dim
constexpr int T  = S * NB;      // 8192 tokens
constexpr float EPS = 1e-5f;
constexpr float INV_TEMP = 0.125f;   // 1/sqrt(DH)

typedef __bf16 bf16_t;
typedef __attribute__((ext_vector_type(16))) __bf16 v16bf;
typedef __attribute__((ext_vector_type(8)))  __bf16 v8bf;
typedef __attribute__((ext_vector_type(8)))  float  v8f;
typedef __attribute__((ext_vector_type(4)))  int    v4i;

#define AS1 __attribute__((address_space(1)))
#define AS3 __attribute__((address_space(3)))

#if defined(__has_builtin)
#  if __has_builtin(__builtin_amdgcn_global_load_async_to_lds_b128)
#    define HAVE_ASYNC_LDS 1
#  else
#    define HAVE_ASYNC_LDS 0
#  endif
#  if __has_builtin(__builtin_amdgcn_s_wait_asynccnt)
#    define HAVE_WAIT_ASYNC 1
#  else
#    define HAVE_WAIT_ASYNC 0
#  endif
#else
#  define HAVE_ASYNC_LDS 0
#  define HAVE_WAIT_ASYNC 0
#endif

__device__ __forceinline__ void wait_async_zero() {
#if HAVE_ASYNC_LDS
#  if HAVE_WAIT_ASYNC
  __builtin_amdgcn_s_wait_asynccnt(0);
#  else
  asm volatile("s_wait_asynccnt 0" ::: "memory");
#  endif
#endif
}

// ---------------------------------------------------------------------------
// Weight convert + transpose: Wt[n][k] = (bf16) W[k][n]   (K-major for WMMA B)
// ---------------------------------------------------------------------------
__global__ __launch_bounds__(256) void wconv_kernel(
    const float* __restrict__ w0, const float* __restrict__ w1,
    const float* __restrict__ w2, const float* __restrict__ w3,
    const float* __restrict__ w4,
    bf16_t* __restrict__ t0, bf16_t* __restrict__ t1,
    bf16_t* __restrict__ t2, bf16_t* __restrict__ t3,
    bf16_t* __restrict__ t4) {
  int idx = blockIdx.x * 256 + threadIdx.x;      // D*D threads
  int k = idx >> 10;
  int n = idx & 1023;
  size_t src = (size_t)k * D + n;                // coalesced read along n
  size_t dst = (size_t)n * D + k;
  t0[dst] = (bf16_t)w0[src];
  t1[dst] = (bf16_t)w1[src];
  t2[dst] = (bf16_t)w2[src];
  t3[dst] = (bf16_t)w3[src];
  t4[dst] = (bf16_t)w4[src];
}

// ---------------------------------------------------------------------------
// (optional residual add) + LayerNorm.  One block per token, 256 threads.
// ---------------------------------------------------------------------------
__global__ __launch_bounds__(256) void add_ln_kernel(
    const float* __restrict__ x, const float* __restrict__ res,
    const float* __restrict__ g, const float* __restrict__ b,
    float* __restrict__ yf, bf16_t* __restrict__ ybf) {
  __shared__ float red[256];
  int t   = blockIdx.x;
  int tid = threadIdx.x;
  const float* xr = x + (size_t)t * D;
  const float* rr = res ? res + (size_t)t * D : nullptr;

  float v[4];
  float s = 0.f;
#pragma unroll
  for (int i = 0; i < 4; ++i) {
    int d = tid + i * 256;
    float val = xr[d];
    if (rr) val += rr[d];
    v[i] = val;
    s += val;
  }
  red[tid] = s;
  __syncthreads();
  for (int off = 128; off > 0; off >>= 1) {
    if (tid < off) red[tid] += red[tid + off];
    __syncthreads();
  }
  float mean = red[0] * (1.f / (float)D);
  __syncthreads();

  float s2 = 0.f;
#pragma unroll
  for (int i = 0; i < 4; ++i) {
    float d0 = v[i] - mean;
    s2 += d0 * d0;
  }
  red[tid] = s2;
  __syncthreads();
  for (int off = 128; off > 0; off >>= 1) {
    if (tid < off) red[tid] += red[tid + off];
    __syncthreads();
  }
  float var  = red[0] * (1.f / (float)D);
  float rstd = rsqrtf(var + EPS);

#pragma unroll
  for (int i = 0; i < 4; ++i) {
    int d = tid + i * 256;
    float yv = (v[i] - mean) * rstd * g[d] + b[d];
    yf[(size_t)t * D + d]  = yv;
    ybf[(size_t)t * D + d] = (bf16_t)yv;
  }
}

// ---------------------------------------------------------------------------
// WMMA GEMM with double-buffered LDS staging (async-to-LDS when available).
//   out[m][n] = sum_k A[m][k] * Wt[n][k]
// Block tile 128(M) x 128(N), K stepped by 32.  256 thr = 8 waves in a
// 4(M)x2(N) grid; each wave owns 2x4 16x16 tiles -> 8 WMMA per k-step.
// ---------------------------------------------------------------------------
template <int MODE>
__global__ __launch_bounds__(256) void gemm_wmma_kernel(
    const bf16_t* __restrict__ A, const bf16_t* __restrict__ Wt,
    const float* __restrict__ bias, const float* __restrict__ res,
    float* __restrict__ outf, bf16_t* __restrict__ outbf) {
  constexpr int K = D;
  constexpr int LDT = 40;                       // LDS row stride (elements)
  __shared__ __align__(32) bf16_t As[2][128][LDT];
  __shared__ __align__(32) bf16_t Bs[2][128][LDT];

  int tid   = threadIdx.x;
  int lane  = tid & 31;
  int wv    = tid >> 5;            // 0..7
  int wm    = wv & 3;              // 0..3 -> 32-row strip
  int wn    = wv >> 2;             // 0..1 -> 64-col strip
  int l16   = lane & 15;
  int lhalf = lane >> 4;

  int m0blk = blockIdx.x * 128;
  int n0blk = blockIdx.y * 128;

  auto stage = [&](int buf, int k) {
#pragma unroll
    for (int i = 0; i < 2; ++i) {
      int c   = tid + i * 256;     // 512 chunks of 16B cover 8KB tile
      int row = c >> 2;
      int ko  = (c & 3) * 8;
      const bf16_t* ga = A  + (size_t)(m0blk + row) * K + k + ko;
      const bf16_t* gb = Wt + (size_t)(n0blk + row) * K + k + ko;
      bf16_t* la = &As[buf][row][ko];
      bf16_t* lb = &Bs[buf][row][ko];
#if HAVE_ASYNC_LDS
      __builtin_amdgcn_global_load_async_to_lds_b128(
          (AS1 v4i*)(v4i*)ga, (AS3 v4i*)(v4i*)la, 0, 0);
      __builtin_amdgcn_global_load_async_to_lds_b128(
          (AS1 v4i*)(v4i*)gb, (AS3 v4i*)(v4i*)lb, 0, 0);
#else
      *(v8bf*)la = *(const v8bf*)ga;
      *(v8bf*)lb = *(const v8bf*)gb;
#endif
    }
  };

  v8f acc[2][4];
#pragma unroll
  for (int mi = 0; mi < 2; ++mi)
#pragma unroll
    for (int ni = 0; ni < 4; ++ni)
#pragma unroll
      for (int r = 0; r < 8; ++r) acc[mi][ni][r] = 0.f;

  constexpr int STEPS = K / 32;
  stage(0, 0);

  for (int kt = 0; kt < STEPS; ++kt) {
    int buf = kt & 1;
    wait_async_zero();
    __syncthreads();
    if (kt + 1 < STEPS) stage(buf ^ 1, (kt + 1) * 32);

    v16bf af[2], bfg[4];
#pragma unroll
    for (int mi = 0; mi < 2; ++mi) {
      int row = wm * 32 + mi * 16 + l16;
      v8bf lo = *(const v8bf*)(&As[buf][row][lhalf * 8]);
      v8bf hi = *(const v8bf*)(&As[buf][row][lhalf * 8 + 16]);
#pragma unroll
      for (int i = 0; i < 8; ++i) { af[mi][i] = lo[i]; af[mi][i + 8] = hi[i]; }
    }
#pragma unroll
    for (int ni = 0; ni < 4; ++ni) {
      int row = wn * 64 + ni * 16 + l16;
      v8bf lo = *(const v8bf*)(&Bs[buf][row][lhalf * 8]);
      v8bf hi = *(const v8bf*)(&Bs[buf][row][lhalf * 8 + 16]);
#pragma unroll
      for (int i = 0; i < 8; ++i) { bfg[ni][i] = lo[i]; bfg[ni][i + 8] = hi[i]; }
    }
#pragma unroll
    for (int mi = 0; mi < 2; ++mi)
#pragma unroll
      for (int ni = 0; ni < 4; ++ni)
        acc[mi][ni] = __builtin_amdgcn_wmma_f32_16x16x32_bf16(
            false, af[mi], false, bfg[ni], (short)0, acc[mi][ni], false, false);
  }

  // Epilogue
#pragma unroll
  for (int mi = 0; mi < 2; ++mi) {
#pragma unroll
    for (int ni = 0; ni < 4; ++ni) {
#pragma unroll
      for (int r = 0; r < 8; ++r) {
        int m = m0blk + wm * 32 + mi * 16 + lhalf * 8 + r;  // token index
        int n = n0blk + wn * 64 + ni * 16 + l16;            // feature index
        float vacc = acc[mi][ni][r];
        if constexpr (MODE == 1) {        // q/k: [nb][h][s][dh] bf16
          int s = m >> 2, nb = m & 3;
          int h = n >> 6, dh = n & 63;
          outbf[(((size_t)(nb * H + h)) * S + s) * DH + dh] = (bf16_t)vacc;
        } else if constexpr (MODE == 2) { // v^T: [nb][h][dh][s] bf16
          int s = m >> 2, nb = m & 3;
          int h = n >> 6, dh = n & 63;
          outbf[(((size_t)(nb * H + h)) * DH + dh) * S + s] = (bf16_t)vacc;
        } else if constexpr (MODE == 3) { // bias + ReLU -> bf16
          float tv = vacc + bias[n];
          outbf[(size_t)m * D + n] = (bf16_t)fmaxf(tv, 0.f);
        } else {                          // MODE 4: bias + residual -> fp32
          outf[(size_t)m * D + n] = res[(size_t)m * D + n] + vacc + bias[n];
        }
      }
    }
  }
}

// ---------------------------------------------------------------------------
// Flash attention. One block = one (nb,h) pair + one 64-row query tile.
// 128 threads = 4 waves; wave w owns the 16-row strip w of the 64x64 tiles.
// ---------------------------------------------------------------------------
__global__ __launch_bounds__(128) void attn_kernel(
    const bf16_t* __restrict__ qr, const bf16_t* __restrict__ kr,
    const bf16_t* __restrict__ vt, float* __restrict__ att) {
  __shared__ float  sS[64][65];
  __shared__ __align__(16) bf16_t pS[64][72];   // 144B rows, 16B aligned
  __shared__ float  mrow[64], lrow[64], crow[64];

  int bid = blockIdx.x;
  int qt  = bid & 31;                // S/64 query tiles
  int nh  = bid >> 5;                // 0..63
  int nb  = nh >> 4;
  int h   = nh & 15;
  int tid   = threadIdx.x;
  int lane  = tid & 31;
  int w     = tid >> 5;              // wave 0..3
  int l16   = lane & 15;
  int lhalf = lane >> 4;

  size_t headQK = ((size_t)(nb * H + h)) * S * DH;   // qr/kr: [s][dh]
  size_t headV  = ((size_t)(nb * H + h)) * DH * S;   // vt:    [dh][s]
  int qs0 = qt * 64;

  if (tid < 64) { mrow[tid] = -1e30f; lrow[tid] = 0.f; }

  // Q A-fragments for this wave's 16-row strip (2 slices of dh)
  v16bf qa[2];
  {
    const bf16_t* qbase = qr + headQK + (size_t)(qs0 + w * 16 + l16) * DH;
#pragma unroll
    for (int ks = 0; ks < 2; ++ks) {
      int k0 = ks * 32 + lhalf * 16;
      v8bf lo = *(const v8bf*)(qbase + k0);
      v8bf hi = *(const v8bf*)(qbase + k0 + 8);
#pragma unroll
      for (int i = 0; i < 8; ++i) { qa[ks][i] = lo[i]; qa[ks][i + 8] = hi[i]; }
    }
  }
  v8f o[4];
#pragma unroll
  for (int ni = 0; ni < 4; ++ni)
#pragma unroll
    for (int r = 0; r < 8; ++r) o[ni][r] = 0.f;
  __syncthreads();

  for (int kt = 0; kt < S / 64; ++kt) {
    int ks0 = kt * 64;
    // Prefetch next key/value tiles toward L2 while computing this one.
    if (kt + 1 < S / 64) {
      __builtin_prefetch(kr + headQK + (size_t)(ks0 + 64 + lane) * DH, 0, 1);
      __builtin_prefetch(vt + headV + (size_t)(lane) * S + ks0 + 64, 0, 1);
    }
    // ---- S tile = Q . K^T (scaled) -> LDS -------------------------------
#pragma unroll
    for (int ni = 0; ni < 4; ++ni) {
      v8f acc;
#pragma unroll
      for (int r = 0; r < 8; ++r) acc[r] = 0.f;
#pragma unroll
      for (int ks = 0; ks < 2; ++ks) {
        int key = ks0 + ni * 16 + l16;
        v16bf bfg = *(const v16bf*)(kr + headQK + (size_t)key * DH + ks * 32 + lhalf * 16);
        acc = __builtin_amdgcn_wmma_f32_16x16x32_bf16(
            false, qa[ks], false, bfg, (short)0, acc, false, false);
      }
      int mbase = w * 16 + lhalf * 8;
#pragma unroll
      for (int r = 0; r < 8; ++r)
        sS[mbase + r][ni * 16 + l16] = acc[r] * INV_TEMP;
    }
    __syncthreads();

    // ---- Online softmax over the 64-key tile ---------------------------
    if (tid < 64) {
      int rr = tid;
      float mprev = mrow[rr];
      float mx = mprev;
      for (int j = 0; j < 64; ++j) mx = fmaxf(mx, sS[rr][j]);
      float corrf = __expf(mprev - mx);
      float lsum = 0.f;
      for (int j = 0; j < 64; ++j) {
        float pv = __expf(sS[rr][j] - mx);
        lsum += pv;
        pS[rr][j] = (bf16_t)pv;
      }
      mrow[rr] = mx;
      lrow[rr] = lrow[rr] * corrf + lsum;
      crow[rr] = corrf;
    }
    __syncthreads();

    // ---- Rescale O and accumulate P . V --------------------------------
    v16bf pa[2];
#pragma unroll
    for (int ks = 0; ks < 2; ++ks) {
      int mm = w * 16 + l16;
      int k0 = ks * 32 + lhalf * 16;
      v8bf lo = *(const v8bf*)(&pS[mm][k0]);
      v8bf hi = *(const v8bf*)(&pS[mm][k0 + 8]);
#pragma unroll
      for (int i = 0; i < 8; ++i) { pa[ks][i] = lo[i]; pa[ks][i + 8] = hi[i]; }
    }
    float cf[8];
    {
      int mbase = w * 16 + lhalf * 8;
#pragma unroll
      for (int r = 0; r < 8; ++r) cf[r] = crow[mbase + r];
    }
#pragma unroll
    for (int ni = 0; ni < 4; ++ni) {
#pragma unroll
      for (int r = 0; r < 8; ++r) o[ni][r] *= cf[r];
#pragma unroll
      for (int ks = 0; ks < 2; ++ks) {
        int dh = ni * 16 + l16;
        v16bf bfg = *(const v16bf*)(vt + headV + (size_t)dh * S + ks0 + ks * 32 + lhalf * 16);
        o[ni] = __builtin_amdgcn_wmma_f32_16x16x32_bf16(
            false, pa[ks], false, bfg, (short)0, o[ni], false, false);
      }
    }
    __syncthreads();
  }

  // ---- Normalize and write att in [S][NB][D] layout ---------------------
  float linv[8];
  {
    int mbase = w * 16 + lhalf * 8;
#pragma unroll
    for (int r = 0; r < 8; ++r) linv[r] = 1.f / lrow[mbase + r];
  }
#pragma unroll
  for (int ni = 0; ni < 4; ++ni) {
#pragma unroll
    for (int r = 0; r < 8; ++r) {
      int m = w * 16 + lhalf * 8 + r;
      int s = qs0 + m;
      int d = h * DH + ni * 16 + l16;
      att[((size_t)s * NB + nb) * D + d] = o[ni][r] * linv[r];
    }
  }
}

// ---------------------------------------------------------------------------
// Host side
// ---------------------------------------------------------------------------
extern "C" void kernel_launch(void* const* d_in, const int* in_sizes, int n_in,
                              void* d_out, int out_size, void* d_ws, size_t ws_size,
                              hipStream_t stream) {
  const float* x   = (const float*)d_in[0];
  const float* Wq  = (const float*)d_in[1];
  const float* Wk  = (const float*)d_in[2];
  const float* Wv  = (const float*)d_in[3];
  const float* g1  = (const float*)d_in[4];
  const float* b1  = (const float*)d_in[5];
  const float* g2  = (const float*)d_in[6];
  const float* b2  = (const float*)d_in[7];
  const float* Wf1 = (const float*)d_in[8];
  const float* bf1 = (const float*)d_in[9];
  const float* Wf2 = (const float*)d_in[10];
  const float* bf2 = (const float*)d_in[11];

  char* p = (char*)d_ws;
  auto carve = [&](size_t bytes) -> void* {
    void* r = (void*)p;
    p += (bytes + 255) & ~(size_t)255;
    return r;
  };
  bf16_t* Wq_t  = (bf16_t*)carve((size_t)D * D * 2);
  bf16_t* Wk_t  = (bf16_t*)carve((size_t)D * D * 2);
  bf16_t* Wv_t  = (bf16_t*)carve((size_t)D * D * 2);
  bf16_t* Wf1_t = (bf16_t*)carve((size_t)D * D * 2);
  bf16_t* Wf2_t = (bf16_t*)carve((size_t)D * D * 2);
  float*  xn_f  = (float*) carve((size_t)T * D * 4);
  bf16_t* xn_b  = (bf16_t*)carve((size_t)T * D * 2);
  bf16_t* q_r   = (bf16_t*)carve((size_t)T * D * 2);
  bf16_t* k_r   = (bf16_t*)carve((size_t)T * D * 2);
  bf16_t* v_t   = (bf16_t*)carve((size_t)T * D * 2);
  float*  att   = (float*) carve((size_t)T * D * 4);
  float*  xt_f  = (float*) carve((size_t)T * D * 4);
  bf16_t* xt_b  = (bf16_t*)carve((size_t)T * D * 2);
  bf16_t* h1    = (bf16_t*)carve((size_t)T * D * 2);

  // 1) weights -> bf16, transposed to K-major
  wconv_kernel<<<(D * D) / 256, 256, 0, stream>>>(
      Wq, Wk, Wv, Wf1, Wf2, Wq_t, Wk_t, Wv_t, Wf1_t, Wf2_t);

  // 2) LN1
  add_ln_kernel<<<T, 256, 0, stream>>>(x, nullptr, g1, b1, xn_f, xn_b);

  // 3) Q/K/V projections (WMMA GEMMs) with head-layout scatter epilogues
  dim3 ggrid(T / 128, D / 128);
  gemm_wmma_kernel<1><<<ggrid, 256, 0, stream>>>(xn_b, Wq_t, nullptr, nullptr, nullptr, q_r);
  gemm_wmma_kernel<1><<<ggrid, 256, 0, stream>>>(xn_b, Wk_t, nullptr, nullptr, nullptr, k_r);
  gemm_wmma_kernel<2><<<ggrid, 256, 0, stream>>>(xn_b, Wv_t, nullptr, nullptr, nullptr, v_t);

  // 4) Flash attention: NB*H*(S/64) = 2048 blocks
  attn_kernel<<<NB * H * (S / 64), 128, 0, stream>>>(q_r, k_r, v_t, att);

  // 5) residual + LN2
  add_ln_kernel<<<T, 256, 0, stream>>>(xn_f, att, g2, b2, xt_f, xt_b);

  // 6) FFN1: bias + ReLU -> bf16
  gemm_wmma_kernel<3><<<ggrid, 256, 0, stream>>>(xt_b, Wf1_t, bf1, nullptr, nullptr, h1);

  // 7) FFN2: bias + residual -> fp32 output
  gemm_wmma_kernel<4><<<ggrid, 256, 0, stream>>>(h1, Wf2_t, bf2, xt_f, (float*)d_out, nullptr);
}
